// MyGRUCell_81174881894903
// MI455X (gfx1250) — matrix operand order repaired and checked by reading, not compiled, not run on hardware
//
#include <hip/hip_runtime.h>

#define B_SZ 16384
#define I_SZ 1024
#define H_SZ 1024

typedef __attribute__((ext_vector_type(16))) _Float16 v16h;
typedef __attribute__((ext_vector_type(8)))  _Float16 v8h;
typedef __attribute__((ext_vector_type(8)))  float    v8f;
typedef __attribute__((ext_vector_type(4)))  float    v4f;

// ---------------------------------------------------------------------------
// fp32 -> fp16 conversion
// ---------------------------------------------------------------------------
__global__ void convert_f32_to_f16(const float* __restrict__ src,
                                   _Float16* __restrict__ dst, int n) {
    int i = (blockIdx.x * blockDim.x + threadIdx.x) * 8;
    if (i < n) {
        v4f a = *(const v4f*)(src + i);
        v4f b = *(const v4f*)(src + i + 4);
        v8h o;
        o[0] = (_Float16)a[0]; o[1] = (_Float16)a[1];
        o[2] = (_Float16)a[2]; o[3] = (_Float16)a[3];
        o[4] = (_Float16)b[0]; o[5] = (_Float16)b[1];
        o[6] = (_Float16)b[2]; o[7] = (_Float16)b[3];
        *(v8h*)(dst + i) = o;
    }
}

// ---------------------------------------------------------------------------
// CDNA5 async DMA: global -> LDS, no data VGPRs, tracked by ASYNCcnt.
//   LDS destination address comes from the VDST VGPR; generic LDS pointers carry the LDS byte
//   offset in addr[31:0] (ISA 10.2 aperture rules).
// ---------------------------------------------------------------------------
__device__ __forceinline__ void async_copy16(unsigned int lds_byte_off,
                                             const void* gptr) {
    unsigned long long ga = (unsigned long long)(uintptr_t)gptr;
    asm volatile("global_load_async_to_lds_b128 %0, %1, off"
                 :: "v"(lds_byte_off), "v"(ga) : "memory");
}
__device__ __forceinline__ void wait_async0() {
    asm volatile("s_wait_asynccnt 0x0" ::: "memory");
}

// ---------------------------------------------------------------------------
// WMMA fragment helpers (layouts per CDNA5 ISA 7.12.2, wave32)
// ---------------------------------------------------------------------------
__device__ __forceinline__ v16h load_a_frag(const _Float16* __restrict__ base,
                                            int row0, int k0, int lane) {
    int hlf = lane >> 4;
    int r   = lane & 15;
    const _Float16* p = base + (row0 + r) * I_SZ + k0 + 8 * hlf;
    v8h lo = *(const v8h*)(p);
    v8h hi = *(const v8h*)(p + 16);
    v16h a;
#pragma unroll
    for (int i = 0; i < 8; ++i) { a[i] = lo[i]; a[8 + i] = hi[i]; }
    return a;
}

__device__ __forceinline__ v16h lds_b_frag(const _Float16* sb, int mat) {
    enum { MAT_HALVES = 32 * 40 };
    v8h lo = *(const v8h*)(sb + mat * MAT_HALVES);
    v8h hi = *(const v8h*)(sb + mat * MAT_HALVES + 16);
    v16h b;
#pragma unroll
    for (int i = 0; i < 8; ++i) { b[i] = lo[i]; b[8 + i] = hi[i]; }
    return b;
}

__device__ __forceinline__ v8f wmma16(v16h a, v16h b, v8f c) {
    return __builtin_amdgcn_wmma_f32_16x16x32_f16(
        false, a, false, b, (short)0, c, false, false);
}

__device__ __forceinline__ float sigmoidf_(float x) {
    return 1.0f / (1.0f + __expf(-x));
}

// ---------------------------------------------------------------------------
// Fused GRU cell; weights staged to LDS by async DMA, double buffered,
// one barrier per K-step.
//   block = 256 threads = 8 waves; wave tile 32(M) x 16(N);
//   wave grid 4(M) x 2(N) => block tile 128 x 32; grid (128, 32).
// ---------------------------------------------------------------------------
#define LDS_STRIDE 40             // halves per padded row (80 B)
#define MAT_LDS    (32 * LDS_STRIDE)
#define BUF_LDS    (6 * MAT_LDS)  // 7680 halves = 15360 B per stage
#define BUF_BYTES  (BUF_LDS * 2)
#define W_STRIDE   (H_SZ * I_SZ)  // halves between consecutive weight matrices
#define KSTEPS     (I_SZ / 32)    // 32

__global__ __launch_bounds__(256) void gru_wmma_kernel(
    const _Float16* __restrict__ Xh,  const _Float16* __restrict__ Hh,
    const _Float16* __restrict__ Wbase,   // [iz, ir, ih, hz, hr, hh] contiguous
    const float* __restrict__ biz, const float* __restrict__ bir,
    const float* __restrict__ bih, const float* __restrict__ bhz,
    const float* __restrict__ bhr, const float* __restrict__ bhh,
    const float* __restrict__ h_prev, float* __restrict__ out)
{
    __shared__ __align__(16) _Float16 smem[2 * BUF_LDS];   // 30720 B

    const int tid  = threadIdx.x;
    const int lane = tid & 31;
    const int wave = tid >> 5;
    const int wm = wave & 3;
    const int wn = wave >> 2;
    const int m0    = blockIdx.x * 128 + wm * 32;
    const int n0blk = blockIdx.y * 32;
    const int n0    = n0blk + wn * 16;

    const int hlf = lane >> 4;
    const int r   = lane & 15;

    const unsigned int smemB = (unsigned int)(uintptr_t)smem;  // LDS byte offset

    // per-thread staging chunk coordinates (3 x 16B per thread, 768 chunks)
    int gOff[3];            // offset into Wbase (halves); add k per step
    unsigned int lOffB[3];  // byte offset within a stage buffer
#pragma unroll
    for (int j = 0; j < 3; ++j) {
        int c    = tid + j * 256;
        int mat  = c >> 7;
        int cc   = c & 127;
        int row  = cc >> 2;
        int part = cc & 3;
        gOff[j]  = mat * W_STRIDE + (n0blk + row) * I_SZ + part * 8;
        lOffB[j] = (unsigned int)(mat * MAT_LDS + row * LDS_STRIDE + part * 8) * 2u;
    }

    const v8f zf = {0.f, 0.f, 0.f, 0.f, 0.f, 0.f, 0.f, 0.f};
    v8f az[2]  = {zf, zf};
    v8f ar[2]  = {zf, zf};
    v8f axh[2] = {zf, zf};
    v8f ahh[2] = {zf, zf};

    // ---- prologue: async-stage K-step 0 into buffer 0 ----------------------
#pragma unroll
    for (int j = 0; j < 3; ++j)
        async_copy16(smemB + lOffB[j], Wbase + gOff[j]);
    wait_async0();

    unsigned int bufB = 0;
#pragma unroll 1
    for (int it = 0; it < KSTEPS; ++it) {
        __syncthreads();          // stage at bufB fully written for all waves

        const int k0 = it * 32;
        const bool more = (it + 1) < KSTEPS;

        // async-stage next K-step into the other buffer (no VGPR data)
        if (more) {
#pragma unroll
            for (int j = 0; j < 3; ++j)
                async_copy16(smemB + (bufB ^ BUF_BYTES) + lOffB[j],
                             Wbase + gOff[j] + k0 + 32);
        }

        const _Float16* sb = smem + (bufB >> 1) +
                             (wn * 16 + r) * LDS_STRIDE + 8 * hlf;

        // ---- X phase: mats {iz, ir, ih} consume ax0/ax1 --------------------
        {
            v16h ax0 = load_a_frag(Xh, m0,      k0, lane);
            v16h ax1 = load_a_frag(Xh, m0 + 16, k0, lane);
            v16h b;
            b = lds_b_frag(sb, 0);
            az[0]  = wmma16(ax0, b, az[0]);   az[1]  = wmma16(ax1, b, az[1]);
            b = lds_b_frag(sb, 1);
            ar[0]  = wmma16(ax0, b, ar[0]);   ar[1]  = wmma16(ax1, b, ar[1]);
            b = lds_b_frag(sb, 2);
            axh[0] = wmma16(ax0, b, axh[0]);  axh[1] = wmma16(ax1, b, axh[1]);
        }
        // ---- H phase: mats {hz, hr, hh} consume ah0/ah1 --------------------
        {
            v16h ah0 = load_a_frag(Hh, m0,      k0, lane);
            v16h ah1 = load_a_frag(Hh, m0 + 16, k0, lane);
            v16h b;
            b = lds_b_frag(sb, 3);
            az[0]  = wmma16(ah0, b, az[0]);   az[1]  = wmma16(ah1, b, az[1]);
            b = lds_b_frag(sb, 4);
            ar[0]  = wmma16(ah0, b, ar[0]);   ar[1]  = wmma16(ah1, b, ar[1]);
            b = lds_b_frag(sb, 5);
            ahh[0] = wmma16(ah0, b, ahh[0]);  ahh[1] = wmma16(ah1, b, ahh[1]);
        }

        // own async copies must be in LDS before we signal the next barrier
        if (more) wait_async0();
        bufB ^= BUF_BYTES;
    }

    // ---- epilogue: C/D layout VGPR v -> M = v + 8*hlf, N = lane&15 ---------
    const int n = n0 + r;
    const float bz = biz[n] + bhz[n];
    const float br = bir[n] + bhr[n];
    const float bx = bih[n];
    const float bh = bhh[n];

#pragma unroll
    for (int s = 0; s < 2; ++s) {
#pragma unroll
        for (int v = 0; v < 8; ++v) {
            const int m = m0 + 16 * s + 8 * hlf + v;
            const float z  = sigmoidf_(az[s][v] + bz);
            const float rr = sigmoidf_(ar[s][v] + br);
            const float g  = tanhf(axh[s][v] + bx + rr * (ahh[s][v] + bh));
            const float hp = h_prev[m * H_SZ + n];
            out[m * H_SZ + n] = (1.0f - z) * g + z * hp;
        }
    }
}

// ---------------------------------------------------------------------------
// Launch
// ---------------------------------------------------------------------------
extern "C" void kernel_launch(void* const* d_in, const int* in_sizes, int n_in,
                              void* d_out, int out_size, void* d_ws, size_t ws_size,
                              hipStream_t stream) {
    const float* x      = (const float*)d_in[0];
    const float* h_prev = (const float*)d_in[1];
    const float* Wiz_w  = (const float*)d_in[2];
    const float* Wiz_b  = (const float*)d_in[3];
    const float* Wir_w  = (const float*)d_in[4];
    const float* Wir_b  = (const float*)d_in[5];
    const float* Wih_w  = (const float*)d_in[6];
    const float* Wih_b  = (const float*)d_in[7];
    const float* Whz_w  = (const float*)d_in[8];
    const float* Whz_b  = (const float*)d_in[9];
    const float* Whr_w  = (const float*)d_in[10];
    const float* Whr_b  = (const float*)d_in[11];
    const float* Whh_w  = (const float*)d_in[12];
    const float* Whh_b  = (const float*)d_in[13];
    float* out = (float*)d_out;

    // ws layout (fp16): Xh | Hh | Wiz | Wir | Wih | Whz | Whr | Whh (contiguous)
    const size_t nX = (size_t)B_SZ * I_SZ;
    const size_t nH = (size_t)B_SZ * H_SZ;
    const size_t nW = (size_t)H_SZ * I_SZ;
    _Float16* Xh = (_Float16*)d_ws;
    _Float16* Hh = Xh + nX;
    _Float16* Wb = Hh + nH;            // 6 matrices, contiguous

    const int cvt_block = 256;
    auto cvt = [&](const float* s, _Float16* d, size_t n) {
        int blocks = (int)(n / (8 * (size_t)cvt_block));
        convert_f32_to_f16<<<blocks, cvt_block, 0, stream>>>(s, d, (int)n);
    };
    cvt(x,      Xh,          nX);
    cvt(h_prev, Hh,          nH);
    cvt(Wiz_w,  Wb + 0 * nW, nW);
    cvt(Wir_w,  Wb + 1 * nW, nW);
    cvt(Wih_w,  Wb + 2 * nW, nW);
    cvt(Whz_w,  Wb + 3 * nW, nW);
    cvt(Whr_w,  Wb + 4 * nW, nW);
    cvt(Whh_w,  Wb + 5 * nW, nW);

    dim3 grid(B_SZ / 128, H_SZ / 32);
    gru_wmma_kernel<<<grid, 256, 0, stream>>>(
        Xh, Hh, Wb,
        Wiz_b, Wir_b, Wih_b, Whz_b, Whr_b, Whh_b,
        h_prev, out);
}
